// GAT_66202625901208
// MI455X (gfx1250) — matrix operand order repaired and checked by reading, not compiled, not run on hardware
//
#include <hip/hip_runtime.h>

// ---------------------------------------------------------------------------
// GAT pipeline for MI455X (gfx1250, wave32, WMMA).
//   MLP GEMMs + GAT feature GEMMs -> v_wmma_f32_16x16x32_f16 with
//   fragment-packed operands (contiguous 32B/lane loads, no repacking VALU)
//   Edge softmax/aggregation      -> atomic passes (max_u32 / add_f32)
// ---------------------------------------------------------------------------

typedef _Float16 v16h __attribute__((ext_vector_type(16)));
typedef float    v8f  __attribute__((ext_vector_type(8)));

// Packed-A/B fragment index for WMMA 16x16x32 f16 (ISA 7.12.2 layouts).
// Element (row n, depth k) of an A-operand with kT = K/32 k-tiles:
__device__ __forceinline__ size_t pIdx(int n, int k, int kT) {
  int mt = n >> 4, lm = n & 15;
  int kt = k >> 5, kin = k & 31;
  int hi = (kin >> 3) & 1;                    // which half-wave holds it
  int j  = (kin & 7) | ((kin >> 4) << 3);     // element within v16h
  return ((((size_t)mt * kT + kt) * 32) + (hi * 16 + lm)) * 16 + j;
}

// ---------------- staging kernels -----------------------------------------

// x (f32, row-major [M,K]) -> packed f16 A fragments
__global__ void k_pack_a_f32(const float* __restrict__ X,
                             _Float16* __restrict__ Ap, int M, int logK) {
  int idx = blockIdx.x * blockDim.x + threadIdx.x;
  const int K = 1 << logK;
  if (idx >= (M << logK)) return;
  int n = idx >> logK, k = idx & (K - 1);
  Ap[pIdx(n, k, K >> 5)] = (_Float16)X[idx];
}

// W (f32, row-major [K,Nc]) -> packed f16 B fragments:
// Wp[((kt*nT+nt)*32+lane)*16+i] = W[kt*32+(i>>3)*16+(lane>>4)*8+(i&7)][nt*16+(lane&15)]
__global__ void k_pack_w(const float* __restrict__ W,
                         _Float16* __restrict__ Wp, int K, int Nc) {
  int idx = blockIdx.x * blockDim.x + threadIdx.x;
  if (idx >= K * Nc) return;
  int i    = idx & 15;
  int lane = (idx >> 4) & 31;
  int rest = idx >> 9;
  int nT   = Nc >> 4;
  int nt   = rest % nT;
  int kt   = rest / nT;
  int k    = (kt << 5) + ((i >> 3) << 4) + ((lane >> 4) << 3) + (i & 7);
  int col  = (nt << 4) + (lane & 15);
  Wp[idx] = (_Float16)W[(size_t)k * Nc + col];
}

__global__ void k_zero_u32(unsigned* __restrict__ p, int count) {
  int i = blockIdx.x * blockDim.x + threadIdx.x;
  if (i < count) p[i] = 0u;
}

// monotone float <-> uint mapping so unsigned atomicMax == float max
__device__ __forceinline__ unsigned encF(float f) {
  unsigned u = __float_as_uint(f);
  return (u & 0x80000000u) ? ~u : (u ^ 0x80000000u);
}
__device__ __forceinline__ float decF(unsigned u) {
  unsigned v = (u & 0x80000000u) ? (u ^ 0x80000000u) : ~u;
  return __uint_as_float(v);
}

__device__ __forceinline__ void edge_sd(const int* __restrict__ ei, int e,
                                        int E, int& s, int& d) {
  if (e < E) { s = ei[e]; d = ei[E + e]; }
  else       { s = e - E; d = e - E; }        // self loops appended
}

// ---------------- WMMA GEMM: C[M,Nc] = A[M,K] * W[K,Nc] -------------------
// Ap, Wp fragment-packed f16.  Each wave: 32-row x 64-col tile
// (2 M-tiles x 4 N-tiles, B fragment reused by both halves).
// relu_f16out=1: outh = relu(acc+bias), fragment-packed f16 for next GEMM
// relu_f16out=0: outf = acc, natural row-major f32
__global__ void k_gemm_wmma(const _Float16* __restrict__ Ap,
                            const _Float16* __restrict__ Wp,
                            const float* __restrict__ bias,
                            _Float16* __restrict__ outh,
                            float* __restrict__ outf,
                            int M, int K, int Nc, int relu_f16out) {
  const int lane   = threadIdx.x & 31;
  const int wave   = (blockIdx.x * blockDim.x + threadIdx.x) >> 5;
  const int nStrip = Nc >> 6;                 // 64-wide strips
  const int mPairs = (M + 31) >> 5;           // 32-row tiles
  if (wave >= mPairs * nStrip) return;
  const int mp    = wave / nStrip;
  const int strip = wave % nStrip;
  const int mt0   = mp * 2;
  const int kT    = K >> 5;
  const int nT    = Nc >> 4;

  v8f acc[2][4] = {};
  for (int kt = 0; kt < kT; ++kt) {
    v16h a0 = *(const v16h*)(Ap + ((((size_t)mt0       * kT) + kt) * 32 + lane) * 16);
    v16h a1 = *(const v16h*)(Ap + ((((size_t)(mt0 + 1) * kT) + kt) * 32 + lane) * 16);
#pragma unroll
    for (int t = 0; t < 4; ++t) {
      const int nt = strip * 4 + t;
      v16h b = *(const v16h*)(Wp + ((((size_t)kt * nT) + nt) * 32 + lane) * 16);
      acc[0][t] = __builtin_amdgcn_wmma_f32_16x16x32_f16(
          false, a0, false, b, (short)0, acc[0][t], false, false);
      acc[1][t] = __builtin_amdgcn_wmma_f32_16x16x32_f16(
          false, a1, false, b, (short)0, acc[1][t], false, false);
    }
  }

  const int lm = lane & 15, hi = lane >> 4;
#pragma unroll
  for (int half = 0; half < 2; ++half) {
    const int mbase = (mt0 + half) << 4;
#pragma unroll
    for (int t = 0; t < 4; ++t) {
      const int col = (strip * 4 + t) * 16 + lm;
      const float bv = bias ? bias[col] : 0.0f;
#pragma unroll
      for (int r = 0; r < 8; ++r) {
        const int rowg = mbase + hi * 8 + r;  // D layout: half-wave = M 8..15
        if (rowg >= M) continue;
        float v = acc[half][t][r] + bv;
        if (relu_f16out) {
          v = v > 0.f ? v : 0.f;
          outh[pIdx(rowg, col, Nc >> 5)] = (_Float16)v;
        } else {
          outf[(size_t)rowg * Nc + col] = v;
        }
      }
    }
  }
}

// ---------------- small scalar GEMM (128 -> 8 projection) -----------------
__global__ void k_gemm_small(const _Float16* __restrict__ Ap,
                             const float* __restrict__ W,
                             float* __restrict__ out, int M, int K, int Nc) {
  int idx = blockIdx.x * blockDim.x + threadIdx.x;
  if (idx >= M * Nc) return;
  int n = idx / Nc, c = idx % Nc;
  const int kT = K >> 5;
  float s = 0.f;
  for (int k = 0; k < K; ++k)
    s += (float)Ap[pIdx(n, k, kT)] * W[(size_t)k * Nc + c];
  out[idx] = s;
}

// ---------------- attention coefficients a_src / a_dst --------------------
__global__ void k_attn_coef(const float* __restrict__ hfeat,
                            const float* __restrict__ att_s,
                            const float* __restrict__ att_d,
                            float* __restrict__ asrc,
                            float* __restrict__ adst,
                            int Nn, int H, int C) {
  int idx = blockIdx.x * blockDim.x + threadIdx.x;
  if (idx >= Nn * H) return;
  int n = idx / H, h = idx % H;
  const float* hp = hfeat + (size_t)n * H * C + (size_t)h * C;
  float ss = 0.f, sd = 0.f;
  for (int c = 0; c < C; ++c) {
    float x = hp[c];
    ss += x * att_s[h * C + c];
    sd += x * att_d[h * C + c];
  }
  asrc[idx] = ss;
  adst[idx] = sd;
}

// ---------------- edge pass 1: e = leakyrelu, segment max -----------------
__global__ void k_edge_p1(const int* __restrict__ ei, int E, int Etot,
                          const float* __restrict__ asrc,
                          const float* __restrict__ adst,
                          float* __restrict__ ebuf,
                          unsigned* __restrict__ menc, int logH) {
  int idx = blockIdx.x * blockDim.x + threadIdx.x;
  const int H = 1 << logH;
  if (idx >= Etot * H) return;
  int e = idx >> logH, h = idx & (H - 1);
  int s, d; edge_sd(ei, e, E, s, d);
  float v = asrc[s * H + h] + adst[d * H + h];
  v = v > 0.f ? v : 0.2f * v;                  // LeakyReLU(0.2)
  ebuf[idx] = v;
  atomicMax(&menc[d * H + h], encF(v));
}

// ---------------- edge pass 2: ex = exp(e-m), segment sum -----------------
__global__ void k_edge_p2(const int* __restrict__ ei, int E, int Etot,
                          float* __restrict__ ebuf,
                          const unsigned* __restrict__ menc,
                          float* __restrict__ denom, int logH) {
  int idx = blockIdx.x * blockDim.x + threadIdx.x;
  const int H = 1 << logH;
  if (idx >= Etot * H) return;
  int e = idx >> logH, h = idx & (H - 1);
  int s, d; edge_sd(ei, e, E, s, d);
  float m = decF(menc[d * H + h]);
  if (!(m > -1e37f && m < 1e37f)) m = 0.f;     // isfinite guard
  float ex = __expf(ebuf[idx] - m);
  ebuf[idx] = ex;
  atomicAdd(&denom[d * H + h], ex);
}

// ---------------- edge pass 3: scatter-add h[src]*alpha -------------------
__global__ void k_edge_p3(const int* __restrict__ ei, int E, int Etot,
                          const float* __restrict__ ebuf,
                          const float* __restrict__ denom,
                          const float* __restrict__ hfeat,
                          float* __restrict__ agg,
                          int logF, int logH, int logC) {
  int tid = blockIdx.x * blockDim.x + threadIdx.x;
  int e = tid >> logF;
  if (e >= Etot) return;
  const int F = 1 << logF;
  const int H = 1 << logH;
  int f = tid & (F - 1);
  int h = f >> logC;
  int s, d; edge_sd(ei, e, E, s, d);
  float ex = ebuf[e * H + h];
  float dn = denom[d * H + h];
  float alpha = ex / fmaxf(dn, 1e-16f);
  atomicAdd(&agg[(size_t)d * F + f], hfeat[(size_t)s * F + f] * alpha);
}

// ---------------- epilogues -----------------------------------------------
// bias + ELU, stored as fragment-packed f16 A operand for the next GEMM
__global__ void k_bias_elu_f16(const float* __restrict__ agg,
                               const float* __restrict__ bias,
                               _Float16* __restrict__ outh, int Nn, int logF) {
  int idx = blockIdx.x * blockDim.x + threadIdx.x;
  const int F = 1 << logF;
  if (idx >= (Nn << logF)) return;
  int n = idx >> logF, f = idx & (F - 1);
  float v = agg[idx] + bias[f];
  v = v > 0.f ? v : (__expf(v) - 1.f);         // ELU
  outh[pIdx(n, f, F >> 5)] = (_Float16)v;
}

__global__ void k_bias_out(const float* __restrict__ agg,
                           const float* __restrict__ bias,
                           float* __restrict__ out, int Nn, int F) {
  int idx = blockIdx.x * blockDim.x + threadIdx.x;
  if (idx >= Nn * F) return;
  out[idx] = agg[idx] + bias[idx % F];
}

// ---------------------------------------------------------------------------
extern "C" void kernel_launch(void* const* d_in, const int* in_sizes, int n_in,
                              void* d_out, int out_size, void* d_ws, size_t ws_size,
                              hipStream_t stream) {
  (void)n_in; (void)out_size; (void)ws_size;
  const float* x    = (const float*)d_in[0];
  const int*   ei   = (const int*)d_in[1];
  const float* w1   = (const float*)d_in[2];  const float* b1  = (const float*)d_in[3];
  const float* w2   = (const float*)d_in[4];  const float* b2  = (const float*)d_in[5];
  const float* w3   = (const float*)d_in[6];  const float* b3  = (const float*)d_in[7];
  const float* wc1  = (const float*)d_in[8];  const float* as1 = (const float*)d_in[9];
  const float* ad1  = (const float*)d_in[10]; const float* bc1 = (const float*)d_in[11];
  const float* wc2  = (const float*)d_in[12]; const float* as2 = (const float*)d_in[13];
  const float* ad2  = (const float*)d_in[14]; const float* bc2 = (const float*)d_in[15];
  const float* wc3  = (const float*)d_in[16]; const float* as3 = (const float*)d_in[17];
  const float* ad3  = (const float*)d_in[18]; const float* bc3 = (const float*)d_in[19];
  float* out = (float*)d_out;

  const int Nn    = in_sizes[0] / 256;         // 50000
  const int E     = in_sizes[1] / 2;           // 800000
  const int Etot  = E + Nn;                    // self loops appended
  const int NnPad = (Nn + 31) & ~31;           // padded rows for 32-row tiles

  // ---- workspace carve-out (256B aligned) ----
  char* ws = (char*)d_ws;
  size_t off = 0;
  auto carve = [&](size_t bytes) -> char* {
    char* p = ws + off;
    off += (bytes + 255) & ~(size_t)255;
    return p;
  };
  _Float16* actA  = (_Float16*)carve((size_t)NnPad * 512 * 2);
  _Float16* actB  = (_Float16*)carve((size_t)NnPad * 512 * 2);
  float*    hfeat = (float*)   carve((size_t)Nn * 256 * 4);
  float*    agg   = (float*)   carve((size_t)Nn * 256 * 4);
  float*    ebuf  = (float*)   carve((size_t)Etot * 4 * 4);
  float*    asrc  = (float*)   carve((size_t)Nn * 4 * 4);
  float*    adst  = (float*)   carve((size_t)Nn * 4 * 4);
  unsigned* menc  = (unsigned*)carve((size_t)Nn * 4 * 4);   // + denom contiguous
  float*    denom = (float*)   carve((size_t)Nn * 4 * 4);
  _Float16* w1h   = (_Float16*)carve((size_t)(131072 + 131072 + 32768 + 32768 + 32768) * 2);
  _Float16* w2h   = w1h + 131072;
  _Float16* w3h   = w2h + 131072;
  _Float16* wc1h  = w3h + 32768;
  _Float16* wc2h  = wc1h + 32768;

  const int TB = 256;
  auto blocks = [&](long n) { return (unsigned)((n + TB - 1) / TB); };

  // ---- staging: pack x and weights into WMMA fragment layouts ----
  k_pack_a_f32<<<blocks((long)Nn * 256), TB, 0, stream>>>(x, actA, Nn, 8);
  k_pack_w<<<blocks(131072), TB, 0, stream>>>(w1, w1h, 256, 512);
  k_pack_w<<<blocks(131072), TB, 0, stream>>>(w2, w2h, 512, 256);
  k_pack_w<<<blocks(32768),  TB, 0, stream>>>(w3, w3h, 256, 128);
  k_pack_w<<<blocks(32768),  TB, 0, stream>>>(wc1, wc1h, 128, 256);
  k_pack_w<<<blocks(32768),  TB, 0, stream>>>(wc2, wc2h, 256, 128);

  auto gemm = [&](const _Float16* A, const _Float16* W, const float* bias,
                  _Float16* oh, float* of, int M, int K, int Nc, int relu16) {
    const int mPairs = (M + 31) >> 5;
    const int nStrip = Nc >> 6;
    const long waves = (long)mPairs * nStrip;
    k_gemm_wmma<<<blocks(waves * 32), TB, 0, stream>>>(A, W, bias, oh, of,
                                                       M, K, Nc, relu16);
  };

  // ---- MLP ----
  gemm(actA, w1h, b1, actB, nullptr, Nn, 256, 512, 1);   // h1 packed f16
  gemm(actB, w2h, b2, actA, nullptr, Nn, 512, 256, 1);   // h2 packed f16
  gemm(actA, w3h, b3, actB, nullptr, Nn, 256, 128, 1);   // h3 packed f16

  // ---- GAT layer runner (layers 1 & 2) ----
  auto gat_layer = [&](const _Float16* inAct, int Kin,
                       const _Float16* Wh, int F, int H, int C,
                       const float* attS, const float* attD, const float* bias,
                       _Float16* outAct, int logF, int logH, int logC) {
    gemm(inAct, Wh, nullptr, nullptr, hfeat, Nn, Kin, F, 0);       // hfeat f32
    k_attn_coef<<<blocks((long)Nn * H), TB, 0, stream>>>(hfeat, attS, attD,
                                                         asrc, adst, Nn, H, C);
    k_zero_u32<<<blocks((long)Nn * 4 * 2), TB, 0, stream>>>(menc, Nn * 4 * 2);
    k_zero_u32<<<blocks((long)Nn * 256), TB, 0, stream>>>((unsigned*)agg, Nn * 256);
    k_edge_p1<<<blocks((long)Etot * H), TB, 0, stream>>>(ei, E, Etot, asrc, adst,
                                                         ebuf, menc, logH);
    k_edge_p2<<<blocks((long)Etot * H), TB, 0, stream>>>(ei, E, Etot, ebuf,
                                                         menc, denom, logH);
    k_edge_p3<<<blocks((long)Etot * F), TB, 0, stream>>>(ei, E, Etot, ebuf, denom,
                                                         hfeat, agg, logF, logH, logC);
    if (outAct)
      k_bias_elu_f16<<<blocks((long)Nn * F), TB, 0, stream>>>(agg, bias, outAct,
                                                              Nn, logF);
  };

  // GAT1: 128 -> 4x64 concat=256, ELU   (h3 in actB -> h4 in actA)
  gat_layer(actB, 128, wc1h, 256, 4, 64, as1, ad1, bc1, actA, 8, 2, 6);
  // GAT2: 256 -> 4x32 concat=128, ELU   (h4 in actA -> h5 in actB)
  gat_layer(actA, 256, wc2h, 128, 4, 32, as2, ad2, bc2, actB, 7, 2, 5);

  // ---- GAT3: 128 -> 1x8, mean(=identity), +bias, no ELU ----
  k_gemm_small<<<blocks((long)Nn * 8), TB, 0, stream>>>(actB, wc3, hfeat, Nn, 128, 8);
  k_attn_coef<<<blocks((long)Nn), TB, 0, stream>>>(hfeat, as3, ad3, asrc, adst, Nn, 1, 8);
  k_zero_u32<<<blocks((long)Nn * 4 * 2), TB, 0, stream>>>(menc, Nn * 4 * 2);
  k_zero_u32<<<blocks((long)Nn * 256), TB, 0, stream>>>((unsigned*)agg, Nn * 256);
  k_edge_p1<<<blocks((long)Etot), TB, 0, stream>>>(ei, E, Etot, asrc, adst, ebuf, menc, 0);
  k_edge_p2<<<blocks((long)Etot), TB, 0, stream>>>(ei, E, Etot, ebuf, menc, denom, 0);
  k_edge_p3<<<blocks((long)Etot * 8), TB, 0, stream>>>(ei, E, Etot, ebuf, denom,
                                                       hfeat, agg, 3, 0, 3);
  k_bias_out<<<blocks((long)Nn * 8), TB, 0, stream>>>(agg, bc3, out, Nn, 8);
}
